// Decoder_71528385347726
// MI455X (gfx1250) — matrix operand order repaired and checked by reading, not compiled, not run on hardware
//
#include <hip/hip_runtime.h>
#include <hip/hip_bf16.h>
#include <stdint.h>

#define VOCAB_ 10000
#define H_ 512
#define K_ 512
#define V_ 512
#define T_ 512
#define N_ 64
#define L_ 256

typedef __attribute__((ext_vector_type(16))) __bf16 v16bf;
typedef __attribute__((ext_vector_type(8)))  float  v8f;

__device__ __forceinline__ unsigned short f2bf(float f) {
    unsigned int u = __float_as_uint(f);
    u = u + 0x7FFFu + ((u >> 16) & 1u);   // round-to-nearest-even
    return (unsigned short)(u >> 16);
}
__device__ __forceinline__ int imin(int a, int b) { return a < b ? a : b; }

union Frag { uint4 q[2]; v16bf v; };

// A fragment: 16x32 bf16, rows m0..m0+15. Per CDNA5 ISA layout:
// lane<16: K=0..7 in v0..3, K=16..23 in v4..7 ; lane>=16: K=8..15, K=24..31.
__device__ __forceinline__ v16bf load_fragA(const unsigned short* A, long lda,
                                            int M, int m0, int k, int lane) {
    int m = imin(m0 + (lane & 15), M - 1);
    const unsigned short* row = A + (long)m * lda + k + ((lane >> 4) << 3);
    Frag f;
    f.q[0] = *(const uint4*)(row);        // 8 bf16
    f.q[1] = *(const uint4*)(row + 16);   // 8 bf16 (K+16 block)
    return f.v;
}

// B fragment: 32x16 bf16 (KxN). B source is row-major (Nn x K), same K layout as A:
// lane<16: K=0..7, K=16..23 ; lane>=16: K=8..15, K=24..31.
__device__ __forceinline__ v16bf load_fragB(const unsigned short* B, long ldb,
                                            int Nn, int n0, int k, int lane) {
    int n = imin(n0 + (lane & 15), Nn - 1);
    const unsigned short* row = B + (long)n * ldb + k + ((lane >> 4) << 3);
    Frag f;
    f.q[0] = *(const uint4*)(row);
    f.q[1] = *(const uint4*)(row + 16);
    return f.v;
}

// ---------------------------------------------------------------------------
// Small-tile GEMM (32x32 per wave) with dual-product + Cin accumulate support.
// Used for the sequential per-step recurrent GEMMs (M=64) where grid
// parallelism matters more than arithmetic intensity.
// C = A*B^T [+ A2*B2^T] [+ Cin] [+ bias0 + bias1]
// ---------------------------------------------------------------------------
__global__ __launch_bounds__(32) void gemm_bt(
    const unsigned short* __restrict__ A, long lda, long strideA,
    const unsigned short* __restrict__ B, long ldb, long strideB,
    const unsigned short* __restrict__ A2, long lda2,
    const unsigned short* __restrict__ B2, long ldb2, int K2,
    const float* __restrict__ Cin, long ldcin, long strideCin,
    const float* __restrict__ bias0, const float* __restrict__ bias1,
    float* __restrict__ Cf, unsigned short* __restrict__ Cb,
    long ldc, long strideC,
    int M, int Nn, int K)
{
    const int lane = threadIdx.x;
    const int n0 = blockIdx.x * 32;
    const int m0 = blockIdx.y * 32;
    const int b  = blockIdx.z;
    A += (long)b * strideA;
    B += (long)b * strideB;
    if (Cin) Cin += (long)b * strideCin;
    const long coff = (long)b * strideC;

    const int nc0 = n0 + (lane & 15);
    const int nc1 = nc0 + 16;
    float bv0 = 0.f, bv1 = 0.f;
    if (bias0) { bv0 += bias0[imin(nc0, Nn - 1)]; bv1 += bias0[imin(nc1, Nn - 1)]; }
    if (bias1) { bv0 += bias1[imin(nc0, Nn - 1)]; bv1 += bias1[imin(nc1, Nn - 1)]; }

    v8f c00, c01, c10, c11;
#pragma unroll
    for (int r = 0; r < 8; ++r) {
        float i00 = bv0, i01 = bv1, i10 = bv0, i11 = bv1;
        if (Cin) {
            int mA = imin(m0 + r + ((lane >> 4) << 3), M - 1);
            int mB = imin(mA + 16, M - 1);
            int q0 = imin(nc0, Nn - 1), q1 = imin(nc1, Nn - 1);
            i00 += Cin[(long)mA * ldcin + q0];
            i01 += Cin[(long)mA * ldcin + q1];
            i10 += Cin[(long)mB * ldcin + q0];
            i11 += Cin[(long)mB * ldcin + q1];
        }
        c00[r] = i00; c01[r] = i01; c10[r] = i10; c11[r] = i11;
    }

    for (int k = 0; k < K; k += 32) {
        v16bf a0 = load_fragA(A, lda, M, m0,      k, lane);
        v16bf a1 = load_fragA(A, lda, M, m0 + 16, k, lane);
        v16bf b0 = load_fragB(B, ldb, Nn, n0,      k, lane);
        v16bf b1 = load_fragB(B, ldb, Nn, n0 + 16, k, lane);
        c00 = __builtin_amdgcn_wmma_f32_16x16x32_bf16(false, a0, false, b0, (short)0, c00, false, false);
        c01 = __builtin_amdgcn_wmma_f32_16x16x32_bf16(false, a0, false, b1, (short)0, c01, false, false);
        c10 = __builtin_amdgcn_wmma_f32_16x16x32_bf16(false, a1, false, b0, (short)0, c10, false, false);
        c11 = __builtin_amdgcn_wmma_f32_16x16x32_bf16(false, a1, false, b1, (short)0, c11, false, false);
    }
    if (A2) {
        for (int k = 0; k < K2; k += 32) {
            v16bf a0 = load_fragA(A2, lda2, M, m0,      k, lane);
            v16bf a1 = load_fragA(A2, lda2, M, m0 + 16, k, lane);
            v16bf b0 = load_fragB(B2, ldb2, Nn, n0,      k, lane);
            v16bf b1 = load_fragB(B2, ldb2, Nn, n0 + 16, k, lane);
            c00 = __builtin_amdgcn_wmma_f32_16x16x32_bf16(false, a0, false, b0, (short)0, c00, false, false);
            c01 = __builtin_amdgcn_wmma_f32_16x16x32_bf16(false, a0, false, b1, (short)0, c01, false, false);
            c10 = __builtin_amdgcn_wmma_f32_16x16x32_bf16(false, a1, false, b0, (short)0, c10, false, false);
            c11 = __builtin_amdgcn_wmma_f32_16x16x32_bf16(false, a1, false, b1, (short)0, c11, false, false);
        }
    }

#pragma unroll
    for (int r = 0; r < 8; ++r) {
        int mA = m0 + r + ((lane >> 4) << 3);
        int mB = mA + 16;
        if (mA < M) {
            long ro = coff + (long)mA * ldc;
            if (nc0 < Nn) { if (Cb) Cb[ro + nc0] = f2bf(c00[r]); else Cf[ro + nc0] = c00[r]; }
            if (nc1 < Nn) { if (Cb) Cb[ro + nc1] = f2bf(c01[r]); else Cf[ro + nc1] = c01[r]; }
        }
        if (mB < M) {
            long ro = coff + (long)mB * ldc;
            if (nc0 < Nn) { if (Cb) Cb[ro + nc0] = f2bf(c10[r]); else Cf[ro + nc0] = c10[r]; }
            if (nc1 < Nn) { if (Cb) Cb[ro + nc1] = f2bf(c11[r]); else Cf[ro + nc1] = c11[r]; }
        }
    }
}

// ---------------------------------------------------------------------------
// Big-tile GEMM (64x64 per wave, 4x4 WMMA fragments, 16 accumulators).
// 32 FLOP/byte of operand traffic -> WMMA-bound instead of L1-bound.
// Used for the large parallel GEMMs (energy, context, z_ih1, vocab proj).
// ---------------------------------------------------------------------------
__global__ __launch_bounds__(32) void gemm_bt64(
    const unsigned short* __restrict__ A, long lda, long strideA,
    const unsigned short* __restrict__ B, long ldb, long strideB,
    const float* __restrict__ bias0, const float* __restrict__ bias1,
    float* __restrict__ Cf, unsigned short* __restrict__ Cb,
    long ldc, long strideC,
    int M, int Nn, int K)
{
    const int lane = threadIdx.x;
    const int n0 = blockIdx.x * 64;
    const int m0 = blockIdx.y * 64;
    const int b  = blockIdx.z;
    A += (long)b * strideA;
    B += (long)b * strideB;
    const long coff = (long)b * strideC;

    float bv[4];
#pragma unroll
    for (int j = 0; j < 4; ++j) {
        int nc = imin(n0 + 16 * j + (lane & 15), Nn - 1);
        float t = 0.f;
        if (bias0) t += bias0[nc];
        if (bias1) t += bias1[nc];
        bv[j] = t;
    }

    v8f acc[4][4];
#pragma unroll
    for (int i = 0; i < 4; ++i)
#pragma unroll
        for (int j = 0; j < 4; ++j)
#pragma unroll
            for (int r = 0; r < 8; ++r) acc[i][j][r] = bv[j];

    for (int k = 0; k < K; k += 32) {
        v16bf a[4], bb[4];
#pragma unroll
        for (int i = 0; i < 4; ++i) a[i] = load_fragA(A, lda, M, m0 + 16 * i, k, lane);
#pragma unroll
        for (int j = 0; j < 4; ++j) bb[j] = load_fragB(B, ldb, Nn, n0 + 16 * j, k, lane);
#pragma unroll
        for (int i = 0; i < 4; ++i)
#pragma unroll
            for (int j = 0; j < 4; ++j)
                acc[i][j] = __builtin_amdgcn_wmma_f32_16x16x32_bf16(
                    false, a[i], false, bb[j], (short)0, acc[i][j], false, false);
    }

#pragma unroll
    for (int i = 0; i < 4; ++i) {
#pragma unroll
        for (int r = 0; r < 8; ++r) {
            int m = m0 + 16 * i + r + ((lane >> 4) << 3);
            if (m < M) {
                long ro = coff + (long)m * ldc;
#pragma unroll
                for (int j = 0; j < 4; ++j) {
                    int nc = n0 + 16 * j + (lane & 15);
                    if (nc < Nn) {
                        if (Cb) Cb[ro + nc] = f2bf(acc[i][j][r]);
                        else    Cf[ro + nc] = acc[i][j][r];
                    }
                }
            }
        }
    }
}

__global__ void cvt_bf16_kernel(const float* __restrict__ src,
                                unsigned short* __restrict__ dst, long n) {
    long i = (long)blockIdx.x * blockDim.x + threadIdx.x;
    long stride = (long)gridDim.x * blockDim.x;
    for (; i < n; i += stride) dst[i] = f2bf(src[i]);
}

// values (T,N,V) -> valT_bf (N,V,T) and h2val right half (n*L+t)*1024+512+v for t<L
__global__ void val_prep_kernel(const float* __restrict__ values,
                                unsigned short* __restrict__ valT,
                                unsigned short* __restrict__ h2val) {
    int bx = blockIdx.x;                 // bx = t*N_ + n
    int t = bx / N_, n = bx % N_;
    const float* src = values + ((long)t * N_ + n) * V_;
    for (int v = threadIdx.x; v < V_; v += blockDim.x) {
        unsigned short bv = f2bf(src[v]);
        valT[((long)n * V_ + v) * T_ + t] = bv;
        if (t < L_) h2val[((long)n * L_ + t) * 1024 + 512 + v] = bv;
    }
}

// ce = emb[text[n,l]] -> inp rows (l*N+n), cols 0..511 (bf16)
__global__ void embed_gather_kernel(const float* __restrict__ emb,
                                    const int* __restrict__ text,
                                    unsigned short* __restrict__ inp) {
    int bx = blockIdx.x;                 // bx = l*N_ + n
    int l = bx / N_, n = bx % N_;
    int tok = text[(long)n * L_ + l];
    const float* src = emb + (long)tok * H_;
    unsigned short* dst = inp + (long)bx * 1024;
    for (int h = threadIdx.x; h < H_; h += blockDim.x) dst[h] = f2bf(src[h]);
}

__global__ void init_state_kernel(float* c1, float* c2,
                                  unsigned short* h1, unsigned short* h2, int n) {
    int i = blockIdx.x * blockDim.x + threadIdx.x;
    if (i < n) { c1[i] = 0.f; c2[i] = 0.f; h1[i] = 0; h2[i] = 0; }
}

// masked softmax over T per (n,l); equals softmax->mask->renorm of reference.
__global__ void softmax_att_kernel(const float* __restrict__ E,
                                   const int* __restrict__ text_lens,
                                   unsigned short* __restrict__ att) {
    int l = blockIdx.x, n = blockIdx.y, tid = threadIdx.x;
    const float* e = E + ((long)n * L_ + l) * T_;
    int len = text_lens[n];
    __shared__ float sdata[256];
    int t0 = tid, t1 = tid + 256;
    float e0 = (t0 < len) ? e[t0] : -3.0e38f;
    float e1 = (t1 < len) ? e[t1] : -3.0e38f;
    sdata[tid] = fmaxf(e0, e1);
    __syncthreads();
    for (int s = 128; s > 0; s >>= 1) {
        if (tid < s) sdata[tid] = fmaxf(sdata[tid], sdata[tid + s]);
        __syncthreads();
    }
    float mx = sdata[0];
    __syncthreads();
    float x0 = (t0 < len) ? __expf(e0 - mx) : 0.f;
    float x1 = (t1 < len) ? __expf(e1 - mx) : 0.f;
    sdata[tid] = x0 + x1;
    __syncthreads();
    for (int s = 128; s > 0; s >>= 1) {
        if (tid < s) sdata[tid] += sdata[tid + s];
        __syncthreads();
    }
    float inv = 1.f / sdata[0];
    unsigned short* a = att + ((long)n * L_ + l) * T_;
    a[t0] = f2bf(x0 * inv);
    a[t1] = f2bf(x1 * inv);
}

__device__ __forceinline__ float sigmoidf_(float x) { return 1.f / (1.f + __expf(-x)); }

// z (64 x 2048) f32, c (64 x 512) f32 state; writes h as bf16 (+ optional h2val row)
__global__ void lstm_cell_kernel(const float* __restrict__ z,
                                 float* __restrict__ c,
                                 unsigned short* __restrict__ h_bf,
                                 unsigned short* __restrict__ h2val, int t) {
    int idx = blockIdx.x * blockDim.x + threadIdx.x;   // < 64*512
    int n = idx >> 9, j = idx & 511;
    const float* zr = z + (long)n * 2048;
    float i_ = zr[j], f_ = zr[512 + j], g_ = zr[1024 + j], o_ = zr[1536 + j];
    float cn = sigmoidf_(f_) * c[idx] + sigmoidf_(i_) * tanhf(g_);
    float hn = sigmoidf_(o_) * tanhf(cn);
    c[idx] = cn;
    unsigned short hb = f2bf(hn);
    h_bf[idx] = hb;
    if (h2val) h2val[((long)n * L_ + t) * 1024 + j] = hb;
}

static void launch_gemm(hipStream_t s,
    const unsigned short* A, long lda, long sA,
    const unsigned short* B, long ldb, long sB,
    const unsigned short* A2, long lda2, const unsigned short* B2, long ldb2, int K2,
    const float* Cin, long ldcin, long sCin,
    const float* b0, const float* b1,
    float* Cf, unsigned short* Cb, long ldc, long sC,
    int M, int Nn, int K, int batch)
{
    dim3 g((Nn + 31) / 32, (M + 31) / 32, batch);
    gemm_bt<<<g, 32, 0, s>>>(A, lda, sA, B, ldb, sB, A2, lda2, B2, ldb2, K2,
                             Cin, ldcin, sCin, b0, b1, Cf, Cb, ldc, sC, M, Nn, K);
}

static void launch_gemm64(hipStream_t s,
    const unsigned short* A, long lda, long sA,
    const unsigned short* B, long ldb, long sB,
    const float* b0, const float* b1,
    float* Cf, unsigned short* Cb, long ldc, long sC,
    int M, int Nn, int K, int batch)
{
    dim3 g((Nn + 63) / 64, (M + 63) / 64, batch);
    gemm_bt64<<<g, 32, 0, s>>>(A, lda, sA, B, ldb, sB, b0, b1, Cf, Cb, ldc, sC, M, Nn, K);
}

extern "C" void kernel_launch(void* const* d_in, const int* in_sizes, int n_in,
                              void* d_out, int out_size, void* d_ws, size_t ws_size,
                              hipStream_t stream) {
    const float* key      = (const float*)d_in[0];
    const float* values   = (const float*)d_in[1];
    const int*   text     = (const int*)d_in[2];
    const int*   textlens = (const int*)d_in[3];
    const float* emb      = (const float*)d_in[4];
    const float* w_ih1    = (const float*)d_in[5];
    const float* w_hh1    = (const float*)d_in[6];
    const float* b_ih1    = (const float*)d_in[7];
    const float* b_hh1    = (const float*)d_in[8];
    const float* w_ih2    = (const float*)d_in[9];
    const float* w_hh2    = (const float*)d_in[10];
    const float* b_ih2    = (const float*)d_in[11];
    const float* b_hh2    = (const float*)d_in[12];
    const float* w_out    = (const float*)d_in[13];
    const float* b_out    = (const float*)d_in[14];
    float* out = (float*)d_out;

    char* p = (char*)d_ws;
    auto alloc = [&](size_t bytes) -> char* {
        char* r = p;
        p += (bytes + 255) & ~(size_t)255;
        return r;
    };
    unsigned short* key_bf   = (unsigned short*)alloc((size_t)T_ * N_ * K_ * 2);
    unsigned short* valT_bf  = (unsigned short*)alloc((size_t)N_ * V_ * T_ * 2);
    unsigned short* inp_bf   = (unsigned short*)alloc((size_t)L_ * N_ * 1024 * 2);
    unsigned short* att_bf   = (unsigned short*)alloc((size_t)N_ * L_ * T_ * 2);
    unsigned short* h2val_bf = (unsigned short*)alloc((size_t)L_ * N_ * 1024 * 2);
    unsigned short* wih1_bf  = (unsigned short*)alloc((size_t)2048 * 1024 * 2);
    unsigned short* whh1_bf  = (unsigned short*)alloc((size_t)2048 * 512 * 2);
    unsigned short* wih2_bf  = (unsigned short*)alloc((size_t)2048 * 512 * 2);
    unsigned short* whh2_bf  = (unsigned short*)alloc((size_t)2048 * 512 * 2);
    unsigned short* wout_bf  = (unsigned short*)alloc((size_t)VOCAB_ * 1024 * 2);
    float* z1pre = (float*)alloc((size_t)L_ * N_ * 2048 * 4);  // also aliases E (used earlier)
    float* Ebuf  = z1pre;                                       // (N,L,T) f32, dead before z1pre written
    float* z1buf = (float*)alloc((size_t)N_ * 2048 * 4);
    float* z2buf = (float*)alloc((size_t)N_ * 2048 * 4);
    float* c1    = (float*)alloc((size_t)N_ * H_ * 4);
    float* c2    = (float*)alloc((size_t)N_ * K_ * 4);
    unsigned short* h1_bf = (unsigned short*)alloc((size_t)N_ * H_ * 2);
    unsigned short* h2_bf = (unsigned short*)alloc((size_t)N_ * K_ * 2);

    // --- bf16 conversions of big static operands ---
    cvt_bf16_kernel<<<4096, 256, 0, stream>>>(key,   key_bf,  (long)T_ * N_ * K_);
    cvt_bf16_kernel<<<2048, 256, 0, stream>>>(w_ih1, wih1_bf, (long)2048 * 1024);
    cvt_bf16_kernel<<<1024, 256, 0, stream>>>(w_hh1, whh1_bf, (long)2048 * 512);
    cvt_bf16_kernel<<<1024, 256, 0, stream>>>(w_ih2, wih2_bf, (long)2048 * 512);
    cvt_bf16_kernel<<<1024, 256, 0, stream>>>(w_hh2, whh2_bf, (long)2048 * 512);
    cvt_bf16_kernel<<<8192, 256, 0, stream>>>(w_out, wout_bf, (long)VOCAB_ * 1024);
    val_prep_kernel<<<T_ * N_, 256, 0, stream>>>(values, valT_bf, h2val_bf);
    embed_gather_kernel<<<L_ * N_, 256, 0, stream>>>(emb, text, inp_bf);
    init_state_kernel<<<(N_ * H_ + 255) / 256, 256, 0, stream>>>(c1, c2, h1_bf, h2_bf, N_ * H_);

    // --- attention, all steps in parallel (independent of recurrence) ---
    // energy: E[n,l,t] = sum_k ce[l,n,k] * key[t,n,k]
    launch_gemm64(stream, inp_bf, (long)N_ * 1024, 1024,
                  key_bf, (long)N_ * K_, K_,
                  nullptr, nullptr,
                  Ebuf, nullptr, T_, (long)L_ * T_,
                  L_, T_, H_, N_);
    softmax_att_kernel<<<dim3(L_, N_), 256, 0, stream>>>(Ebuf, textlens, att_bf);
    // context -> inp cols 512..1023 (bf16): ctx[l,n,v] = sum_t att[n,l,t]*valT[n,v,t]
    launch_gemm64(stream, att_bf, T_, (long)L_ * T_,
                  valT_bf, T_, (long)V_ * T_,
                  nullptr, nullptr,
                  nullptr, inp_bf + 512, (long)N_ * 1024, 1024,
                  L_, V_, T_, N_);

    // --- z_ih1 for all steps: (L*N,1024) @ w_ih1^T + b_ih1 + b_hh1 ---
    launch_gemm64(stream, inp_bf, 1024, 0,
                  wih1_bf, 1024, 0,
                  b_ih1, b_hh1,
                  z1pre, nullptr, 2048, 0,
                  L_ * N_, 2048, 1024, 1);

    // --- sequential scan: only recurrent GEMMs remain per step ---
    for (int t = 0; t < L_; ++t) {
        // z1 = z1pre[t] + h1 @ w_hh1^T
        launch_gemm(stream, h1_bf, 512, 0, whh1_bf, 512, 0,
                    nullptr, 0, nullptr, 0, 0,
                    z1pre + (long)t * N_ * 2048, 2048, 0, nullptr, nullptr,
                    z1buf, nullptr, 2048, 0, N_, 2048, 512, 1);
        lstm_cell_kernel<<<(N_ * 512) / 256, 256, 0, stream>>>(z1buf, c1, h1_bf, nullptr, 0);
        // z2 = b_ih2 + b_hh2 + h1 @ w_ih2^T + h2 @ w_hh2^T (fused dual-GEMM)
        launch_gemm(stream, h1_bf, 512, 0, wih2_bf, 512, 0,
                    h2_bf, 512, whh2_bf, 512, 512,
                    nullptr, 0, 0, b_ih2, b_hh2,
                    z2buf, nullptr, 2048, 0, N_, 2048, 512, 1);
        lstm_cell_kernel<<<(N_ * 512) / 256, 256, 0, stream>>>(z2buf, c2, h2_bf, h2val_bf, t);
    }

    // --- output projection, all steps: (L*N,1024) @ w_out^T + b_out -> (N,L,VOCAB) ---
    launch_gemm64(stream, h2val_bf, 1024, 0,
                  wout_bf, 1024, 0,
                  b_out, nullptr,
                  out, nullptr, VOCAB_, 0,
                  L_ * N_, VOCAB_, 1024, 1);
}
